// DeformableBottleneck_85023172592549
// MI455X (gfx1250) — compile-verified, matrix operand order checked by reading, not run on hardware
//
#include <hip/hip_runtime.h>
#include <hip/hip_bf16.h>
#include <stdint.h>

typedef __bf16 bf16_t;
typedef bf16_t v16bf __attribute__((ext_vector_type(16)));
typedef bf16_t v8bf  __attribute__((ext_vector_type(8)));
typedef float  v8f   __attribute__((ext_vector_type(8)));

#define DEVFN __device__ __forceinline__

constexpr int Bn  = 4;
constexpr int CIN = 1024;
constexpr int CB  = 256;
constexpr int H   = 64;
constexpr int W   = 64;
constexpr int HW  = H * W;        // 4096
constexpr int KD  = 9 * CB;       // 2304
constexpr int KQ  = KD / 4;       // 576

DEVFN bf16_t f2bf(float f) {
    unsigned u = __builtin_bit_cast(unsigned, f);
    unsigned r = u + 0x7FFFu + ((u >> 16) & 1u);   // round-to-nearest-even
    unsigned short h = (unsigned short)(r >> 16);
    return __builtin_bit_cast(bf16_t, h);
}

DEVFN float relu(float v) { return v > 0.f ? v : 0.f; }

// WMMA 16-bit A-matrix fragment layout (ISA 7.12.2):
//  lanes 0-15  : M=lane,    slots 0-7 -> K 0..7,  slots 8-15 -> K 16..23
//  lanes 16-31 : M=lane-16, slots 0-7 -> K 8..15, slots 8-15 -> K 24..31
DEVFN int frag_off(int m, int kk) {
    int lane = m + ((kk >> 3) & 1) * 16;
    int slot = (kk & 7) + ((kk & 16) ? 8 : 0);
    return lane * 16 + slot;
}

DEVFN v16bf cat8(v8bf lo, v8bf hi) {
    return __builtin_shufflevector(lo, hi, 0,1,2,3,4,5,6,7,8,9,10,11,12,13,14,15);
}

// Load one B (or direct-A) fragment from a row-major bf16 matrix [rows][Kstride]:
// lane l covers row (base + (l&15)); K chunks at koff and koff+16.
DEVFN v16bf load_frag_rowmajor(const bf16_t* __restrict__ p, int row, int Kstride, int koff) {
    const bf16_t* q = p + (size_t)row * Kstride + koff;
    v8bf lo = *(const v8bf*)(q);
    v8bf hi = *(const v8bf*)(q + 16);
    return cat8(lo, hi);
}

DEVFN v16bf wmma_bf16(v16bf a, v16bf b, v16bf) = delete; // (unused guard)

// ---------------------------------------------------------------------------
// Weight conversion: fp32 -> bf16 (conv2_w also transposed [o][c][t]->[o][t*256+c])
// ---------------------------------------------------------------------------
__global__ void __launch_bounds__(256)
convert_weights(const float* __restrict__ w1, const float* __restrict__ w2,
                const float* __restrict__ w3,
                bf16_t* __restrict__ w1b, bf16_t* __restrict__ w2b,
                bf16_t* __restrict__ w3b) {
    int i = blockIdx.x * 256 + threadIdx.x;
    const int N1 = CB * CIN;       // 262144
    const int N2 = CB * KD;        // 589824
    const int N3 = CIN * CB;       // 262144
    if (i < N1) { w1b[i] = f2bf(w1[i]); return; }
    i -= N1;
    if (i < N2) {
        int n = i / KD, r = i % KD;
        int t = r >> 8, c = r & 255;
        w2b[i] = f2bf(w2[(n * CB + c) * 9 + t]);
        return;
    }
    i -= N2;
    if (i < N3) w3b[i] = f2bf(w3[i]);
}

// ---------------------------------------------------------------------------
// conv1: 1x1 GEMM  M=16384 (pixels) x N=256 x K=1024, BN1+ReLU fused,
// output fp32 NCHW (needed by offset conv + bilinear sampler).
// Block = 128 threads (4 waves); tile M=32, N=256.
// Per wave: 8 accumulators (2 M-halves x 4 N-subtiles); each B fragment is
// reused by two WMMAs. K staged into LDS in two 512-wide halves (32 KB).
// ---------------------------------------------------------------------------
__global__ void __launch_bounds__(128)
conv1_gemm(const float* __restrict__ x, const bf16_t* __restrict__ w1,
           const float* __restrict__ s1, const float* __restrict__ b1,
           float* __restrict__ out1) {
    // [kc(16)][mhalf(2)][512], fragment-ordered
    __shared__ __align__(32) bf16_t ldsA[16 * 2 * 512];   // 32 KB
    const int tid  = threadIdx.x;
    const int lane = tid & 31;
    const int wv   = tid >> 5;
    const int m0   = blockIdx.x * 32;
    const int b    = m0 >> 12;
    const int p0   = m0 & (HW - 1);
    const int nb   = wv * 64;

#pragma unroll
    for (int j = 0; j < 4; ++j)
        __builtin_prefetch(w1 + (size_t)(nb + j * 16 + (lane & 15)) * CIN, 0, 1);

    v8f acc[8] = {};
    for (int half = 0; half < 2; ++half) {
        __syncthreads();
        // Stage + convert A tile (32 pixels x 512 ch) into fragment order
        for (int e = tid; e < 32 * 512; e += 128) {
            int m  = e & 31;
            int kl = e >> 5;
            int k  = half * 512 + kl;
            float v = x[((size_t)(b * CIN + k) * HW) + p0 + m];
            ldsA[(kl >> 5) * 1024 + (m >> 4) * 512 + frag_off(m & 15, kl & 31)] = f2bf(v);
        }
        __syncthreads();

        for (int kc = 0; kc < 16; ++kc) {
            v16bf a0 = *(const v16bf*)(ldsA + kc * 1024 + lane * 16);
            v16bf a1 = *(const v16bf*)(ldsA + kc * 1024 + 512 + lane * 16);
            int koff = half * 512 + kc * 32 + ((lane >> 4) & 1) * 8;
#pragma unroll
            for (int j = 0; j < 4; ++j) {
                int n = nb + j * 16 + (lane & 15);
                v16bf bb = load_frag_rowmajor(w1, n, CIN, koff);
                acc[j]     = __builtin_amdgcn_wmma_f32_16x16x32_bf16(
                    false, a0, false, bb, (short)0, acc[j],     false, false);
                acc[4 + j] = __builtin_amdgcn_wmma_f32_16x16x32_bf16(
                    false, a1, false, bb, (short)0, acc[4 + j], false, false);
            }
        }
    }

#pragma unroll
    for (int hh = 0; hh < 2; ++hh) {
#pragma unroll
        for (int j = 0; j < 4; ++j) {
            int n = nb + j * 16 + (lane & 15);
            float s = s1[n], bo = b1[n];
#pragma unroll
            for (int r = 0; r < 8; ++r) {
                int mloc = hh * 16 + ((lane >> 4) << 3) + r;
                out1[(size_t)(b * CB + n) * HW + p0 + mloc] =
                    relu(acc[hh * 4 + j][r] * s + bo);
            }
        }
    }
}

// ---------------------------------------------------------------------------
// Offset conv: 3x3, 256 -> 18, pad 1.  Small N -> plain VALU kernel.
// ---------------------------------------------------------------------------
__global__ void __launch_bounds__(256)
offset_conv(const float* __restrict__ out1, const float* __restrict__ ow,
            const float* __restrict__ ob, float* __restrict__ offs) {
    int idx = blockIdx.x * 256 + threadIdx.x;
    if (idx >= Bn * 18 * HW) return;
    int p = idx & (HW - 1);
    int o = (idx >> 12) % 18;
    int b = idx / (18 * HW);
    int h = p >> 6, w = p & 63;
    float acc = ob[o];
    for (int c = 0; c < CB; ++c) {
        const float* src = out1 + (size_t)(b * CB + c) * HW;
        const float* wr  = ow + (o * CB + c) * 9;
#pragma unroll
        for (int ty = 0; ty < 3; ++ty) {
            int y = h + ty - 1;
            if (y < 0 || y >= H) continue;
#pragma unroll
            for (int tx = 0; tx < 3; ++tx) {
                int xx = w + tx - 1;
                if (xx < 0 || xx >= W) continue;
                acc += wr[ty * 3 + tx] * src[y * W + xx];
            }
        }
    }
    offs[idx] = acc;
}

// ---------------------------------------------------------------------------
// Deformable conv: bilinear-sample 32x2304 patch into LDS (fragment order,
// four 576-K quarters -> 36 KB), then WMMA GEMM M=32 x N=256 x K=2304 per
// block with B-fragment reuse across the two M-halves. BN2+ReLU fused;
// output bf16 [m][k] row-major so conv3 loads A fragments from global.
// ---------------------------------------------------------------------------
__global__ void __launch_bounds__(128)
deform_gemm(const float* __restrict__ out1, const float* __restrict__ offs,
            const bf16_t* __restrict__ w2, const float* __restrict__ s2,
            const float* __restrict__ b2, bf16_t* __restrict__ out2) {
    // [kc(18)][mhalf(2)][512]
    __shared__ __align__(32) bf16_t ldsA[18 * 2 * 512];   // 36 KB
    __shared__ float pw[288][4];
    __shared__ int   pidx[288][4];
    const int tid  = threadIdx.x;
    const int lane = tid & 31;
    const int wv   = tid >> 5;
    const int m0   = blockIdx.x * 32;
    const int b    = m0 >> 12;
    const int p0   = m0 & (HW - 1);
    const int nb   = wv * 64;

    // Bilinear params for all 32 pixels x 9 taps
    for (int i = tid; i < 288; i += 128) {
        int m = i / 9, t = i % 9;
        int p = p0 + m, h = p >> 6, w = p & 63;
        float dy = offs[((size_t)(b * 18 + 2 * t)     * HW) + p];
        float dx = offs[((size_t)(b * 18 + 2 * t + 1) * HW) + p];
        float ys = dy + (float)(h - 1 + t / 3);
        float xs = dx + (float)(w - 1 + t % 3);
        float y0 = floorf(ys), x0 = floorf(xs);
        float fy = ys - y0, fx = xs - x0;
        int iy0 = (int)y0, ix0 = (int)x0;
#pragma unroll
        for (int q = 0; q < 4; ++q) {
            int yy = iy0 + (q >> 1), xx = ix0 + (q & 1);
            bool valid = (yy >= 0) && (yy < H) && (xx >= 0) && (xx < W);
            float wy = (q >> 1) ? fy : 1.f - fy;
            float wx = (q & 1) ? fx : 1.f - fx;
            int yc = yy < 0 ? 0 : (yy > H - 1 ? H - 1 : yy);
            int xc = xx < 0 ? 0 : (xx > W - 1 ? W - 1 : xx);
            pw[i][q]   = valid ? wy * wx : 0.f;
            pidx[i][q] = yc * W + xc;
        }
    }

#pragma unroll
    for (int j = 0; j < 4; ++j)
        __builtin_prefetch(w2 + (size_t)(nb + j * 16 + (lane & 15)) * KD, 0, 1);

    v8f acc[8] = {};
    for (int quarter = 0; quarter < 4; ++quarter) {
        __syncthreads();   // params ready (1st) / previous GEMM done reading (rest)
        for (int e = tid; e < 32 * KQ; e += 128) {
            int m  = e & 31;
            int kl = e >> 5;
            int k  = quarter * KQ + kl;
            int t  = k >> 8, c = k & 255, mt = m * 9 + t;
            const float* src = out1 + (size_t)(b * CB + c) * HW;
            float v = pw[mt][0] * src[pidx[mt][0]] + pw[mt][1] * src[pidx[mt][1]]
                    + pw[mt][2] * src[pidx[mt][2]] + pw[mt][3] * src[pidx[mt][3]];
            ldsA[(kl >> 5) * 1024 + (m >> 4) * 512 + frag_off(m & 15, kl & 31)] = f2bf(v);
        }
        __syncthreads();

        for (int kc = 0; kc < KQ / 32; ++kc) {
            v16bf a0 = *(const v16bf*)(ldsA + kc * 1024 + lane * 16);
            v16bf a1 = *(const v16bf*)(ldsA + kc * 1024 + 512 + lane * 16);
            int koff = quarter * KQ + kc * 32 + ((lane >> 4) & 1) * 8;
#pragma unroll
            for (int j = 0; j < 4; ++j) {
                int n = nb + j * 16 + (lane & 15);
                v16bf bb = load_frag_rowmajor(w2, n, KD, koff);
                acc[j]     = __builtin_amdgcn_wmma_f32_16x16x32_bf16(
                    false, a0, false, bb, (short)0, acc[j],     false, false);
                acc[4 + j] = __builtin_amdgcn_wmma_f32_16x16x32_bf16(
                    false, a1, false, bb, (short)0, acc[4 + j], false, false);
            }
        }
    }

#pragma unroll
    for (int hh = 0; hh < 2; ++hh) {
#pragma unroll
        for (int j = 0; j < 4; ++j) {
            int n = nb + j * 16 + (lane & 15);
            float s = s2[n], bo = b2[n];
#pragma unroll
            for (int r = 0; r < 8; ++r) {
                int m = m0 + hh * 16 + ((lane >> 4) << 3) + r;
                out2[(size_t)m * CB + n] = f2bf(relu(acc[hh * 4 + j][r] * s + bo));
            }
        }
    }
}

// ---------------------------------------------------------------------------
// conv3: 1x1 GEMM  M=16384 x N=1024 x K=256, A (bf16 row-major) fragments
// loaded directly from global — no LDS. Tile M=32: two A fragments per lane,
// each B fragment reused twice. BN3 + residual(x) + ReLU fused.
// Grid: (M/32, N/256).
// ---------------------------------------------------------------------------
__global__ void __launch_bounds__(128)
conv3_gemm(const bf16_t* __restrict__ a2, const bf16_t* __restrict__ w3,
           const float* __restrict__ s3, const float* __restrict__ b3,
           const float* __restrict__ x, float* __restrict__ out) {
    const int tid  = threadIdx.x;
    const int lane = tid & 31;
    const int wv   = tid >> 5;
    const int m0   = blockIdx.x * 32;
    const int n0   = blockIdx.y * 256;
    const int b    = m0 >> 12;
    const int p0   = m0 & (HW - 1);
    const int nb   = n0 + wv * 64;
    const int mrow = m0 + (lane & 15);

#pragma unroll
    for (int j = 0; j < 4; ++j)
        __builtin_prefetch(w3 + (size_t)(nb + j * 16 + (lane & 15)) * CB, 0, 1);

    v8f acc[8] = {};
    for (int kc = 0; kc < CB / 32; ++kc) {
        int koff = kc * 32 + ((lane >> 4) & 1) * 8;
        v16bf a0 = load_frag_rowmajor(a2, mrow,      CB, koff);
        v16bf a1 = load_frag_rowmajor(a2, mrow + 16, CB, koff);
#pragma unroll
        for (int j = 0; j < 4; ++j) {
            int n = nb + j * 16 + (lane & 15);
            v16bf bb = load_frag_rowmajor(w3, n, CB, koff);
            acc[j]     = __builtin_amdgcn_wmma_f32_16x16x32_bf16(
                false, a0, false, bb, (short)0, acc[j],     false, false);
            acc[4 + j] = __builtin_amdgcn_wmma_f32_16x16x32_bf16(
                false, a1, false, bb, (short)0, acc[4 + j], false, false);
        }
    }

#pragma unroll
    for (int hh = 0; hh < 2; ++hh) {
#pragma unroll
        for (int j = 0; j < 4; ++j) {
            int n = nb + j * 16 + (lane & 15);
            float s = s3[n], bo = b3[n];
#pragma unroll
            for (int r = 0; r < 8; ++r) {
                int mloc = hh * 16 + ((lane >> 4) << 3) + r;
                size_t idx = (size_t)(b * CIN + n) * HW + p0 + mloc;
                out[idx] = relu(acc[hh * 4 + j][r] * s + bo + x[idx]);
            }
        }
    }
}

// ---------------------------------------------------------------------------
extern "C" void kernel_launch(void* const* d_in, const int* in_sizes, int n_in,
                              void* d_out, int out_size, void* d_ws, size_t ws_size,
                              hipStream_t stream) {
    (void)in_sizes; (void)n_in; (void)out_size; (void)ws_size;

    const float* x       = (const float*)d_in[0];
    const float* conv1_w = (const float*)d_in[1];
    const float* bn1_s   = (const float*)d_in[2];
    const float* bn1_b   = (const float*)d_in[3];
    const float* off_w   = (const float*)d_in[4];
    const float* off_b   = (const float*)d_in[5];
    const float* conv2_w = (const float*)d_in[6];
    const float* bn2_s   = (const float*)d_in[7];
    const float* bn2_b   = (const float*)d_in[8];
    const float* conv3_w = (const float*)d_in[9];
    const float* bn3_s   = (const float*)d_in[10];
    const float* bn3_b   = (const float*)d_in[11];
    float* out = (float*)d_out;

    // Workspace carve (256-byte aligned)
    char* ws = (char*)d_ws;
    auto carve = [&](size_t bytes) -> char* {
        char* p = ws;
        ws += (bytes + 255) & ~(size_t)255;
        return p;
    };
    float*  out1 = (float*) carve((size_t)Bn * CB * HW * sizeof(float));   // 16 MB
    float*  offs = (float*) carve((size_t)Bn * 18 * HW * sizeof(float));   // 1.2 MB
    bf16_t* w1b  = (bf16_t*)carve((size_t)CB * CIN * sizeof(bf16_t));      // 0.5 MB
    bf16_t* w2b  = (bf16_t*)carve((size_t)CB * KD  * sizeof(bf16_t));      // 1.2 MB
    bf16_t* w3b  = (bf16_t*)carve((size_t)CIN * CB * sizeof(bf16_t));      // 0.5 MB
    bf16_t* out2 = (bf16_t*)carve((size_t)Bn * HW * CB * sizeof(bf16_t));  // 8 MB

    const int M = Bn * HW;                         // 16384 pixels
    const int convN = CB * CIN + CB * KD + CIN * CB;

    convert_weights<<<(convN + 255) / 256, 256, 0, stream>>>(
        conv1_w, conv2_w, conv3_w, w1b, w2b, w3b);

    conv1_gemm<<<M / 32, 128, 0, stream>>>(x, w1b, bn1_s, bn1_b, out1);

    offset_conv<<<(Bn * 18 * HW + 255) / 256, 256, 0, stream>>>(
        out1, off_w, off_b, offs);

    deform_gemm<<<M / 32, 128, 0, stream>>>(out1, offs, w2b, bn2_s, bn2_b, out2);

    conv3_gemm<<<dim3(M / 32, CIN / 256), 128, 0, stream>>>(
        out2, w3b, bn3_s, bn3_b, x, out);
}
